// EISANImodel_68547678044636
// MI455X (gfx1250) — matrix operand order repaired and checked by reading, not compiled, not run on hardware
//
#include <hip/hip_runtime.h>
#include <hip/hip_bf16.h>

typedef __attribute__((ext_vector_type(16))) _Float16 v16h;
typedef __attribute__((ext_vector_type(8)))  _Float16 h8;
typedef __attribute__((ext_vector_type(8)))  float    v8f;

#define NUM_BITS 16
#define THETA    8.0f
#define HID      4096
#define FEAT     128
#define NCLS     10
#define BATCH    64

// ---------------------------------------------------------------------------
// Thermometer encoding: bits[b, f*16+j] = (x[b,f] >= (j+1)/17) as f16 (0/1).
// ---------------------------------------------------------------------------
__global__ __launch_bounds__(256) void eisani_encode(const float* __restrict__ x,
                                                     _Float16* __restrict__ bits) {
    int i = blockIdx.x * 256 + threadIdx.x;          // over 64*128*16 = 131072
    if (i >= BATCH * FEAT * NUM_BITS) return;
    int j = i & (NUM_BITS - 1);
    int bf = i >> 4;                                  // b*FEAT + f
    float thr = (float)(j + 1) / (float)(NUM_BITS + 1);
    bits[i] = (x[bf] >= thr) ? (_Float16)1.0f : (_Float16)0.0f;
}

// ---------------------------------------------------------------------------
// One EISANI layer: z[b,hs] = sum_p A[b,p] * W[hs,p]   (W row-major [2H, P])
// fired[b,h] = (z[b,2h] >= 8) || (z[b,2h+1] >= 8) ; actsOut f16 0/1 [64, H]
// One wave computes a 16x16 tile of z via v_wmma_f32_16x16x32_f16.
//   grid = (2H/16/8, 64/16), block = 256 (8 waves)
// ---------------------------------------------------------------------------
__global__ __launch_bounds__(256) void eisani_layer(const _Float16* __restrict__ A,
                                                    const float* __restrict__ W,
                                                    _Float16* __restrict__ actsOut,
                                                    int P, int H) {
    const int lane   = threadIdx.x & 31;
    const int wave   = threadIdx.x >> 5;
    const int hsBase = (blockIdx.x * 8 + wave) * 16;   // N-tile base over 2H
    const int mBase  = blockIdx.y * 16;                // M-tile base over batch
    const int lane16 = lane & 15;
    const bool hiHalf = (lane >= 16);

    // A-operand (16x32 f16, M x K): lane = row M; lo half-wave K 0..7 & 16..23,
    // hi half-wave K 8..15 & 24..31 (ISA 16-bit A layout).
    const _Float16* arow = A + (size_t)(mBase + lane16) * P + (hiHalf ? 8 : 0);
    // B-operand (32x16 f16, K x N): lane = column N = one W row; lo half-wave
    // K 0..7 & 16..23, hi half-wave K 8..15 & 24..31 (matches A). f32 -> f16 exact.
    const float* wrow = W + (size_t)(hsBase + lane16) * P + (hiHalf ? 8 : 0);

    v8f c = {};
    for (int k = 0; k < P; k += 32) {
        h8 alo = *(const h8*)(arow + k);
        h8 ahi = *(const h8*)(arow + k + 16);
        v16h a = __builtin_shufflevector(alo, ahi,
                                         0, 1, 2, 3, 4, 5, 6, 7,
                                         8, 9, 10, 11, 12, 13, 14, 15);
        const float4* w4 = (const float4*)(wrow + k);   // two 32-byte runs, 64 bytes apart
        v16h b;
#pragma unroll
        for (int i = 0; i < 4; ++i) {
            float4 f = w4[(i & 1) + (i >> 1) * 4];
            b[4 * i + 0] = (_Float16)f.x;
            b[4 * i + 1] = (_Float16)f.y;
            b[4 * i + 2] = (_Float16)f.z;
            b[4 * i + 3] = (_Float16)f.w;
        }
        c = __builtin_amdgcn_wmma_f32_16x16x32_f16(
                /*neg_a=*/false, a, /*neg_b=*/false, b,
                /*c_mod=*/(short)0, c, /*reuse_a=*/false, /*reuse_b=*/false);
    }

    // C layout: VGPR r holds M = r (lanes 0-15) / r+8 (lanes 16-31), N = lane%16.
    // Segment pair (s=0,1) of neuron h sits at adjacent columns -> lane^1.
    float zx[8];
#pragma unroll
    for (int r = 0; r < 8; ++r) zx[r] = __shfl_xor(c[r], 1, 32);

    if ((lane & 1) == 0) {
        int h  = (hsBase >> 1) + (lane16 >> 1);
        int m0 = mBase + (hiHalf ? 8 : 0);
#pragma unroll
        for (int r = 0; r < 8; ++r) {
            bool fired = (c[r] >= THETA) || (zx[r] >= THETA);
            actsOut[(size_t)(m0 + r) * H + h] = fired ? (_Float16)1.0f
                                                      : (_Float16)0.0f;
        }
    }
}

// ---------------------------------------------------------------------------
// out[b,c] = sum_l sum_h acts[l,b,h] * outW[l,h,c]   grid = (10, 64)
// ---------------------------------------------------------------------------
__global__ __launch_bounds__(256) void eisani_out(const _Float16* __restrict__ acts,
                                                  const float* __restrict__ outW,
                                                  float* __restrict__ out) {
    __shared__ float red[256];
    const int c = blockIdx.x;
    const int b = blockIdx.y;
    float p = 0.0f;
    for (int i = threadIdx.x; i < 3 * HID; i += 256) {
        int l = i >> 12;            // /4096
        int h = i & (HID - 1);
        p += (float)acts[(size_t)l * BATCH * HID + (size_t)b * HID + h] *
             outW[(size_t)l * HID * NCLS + (size_t)h * NCLS + c];
    }
    red[threadIdx.x] = p;
    __syncthreads();
    for (int s = 128; s > 0; s >>= 1) {
        if (threadIdx.x < s) red[threadIdx.x] += red[threadIdx.x + s];
        __syncthreads();
    }
    if (threadIdx.x == 0) out[b * NCLS + c] = red[0];
}

// ---------------------------------------------------------------------------
extern "C" void kernel_launch(void* const* d_in, const int* in_sizes, int n_in,
                              void* d_out, int out_size, void* d_ws, size_t ws_size,
                              hipStream_t stream) {
    const float* x    = (const float*)d_in[0];   // [64,128]
    const float* W0   = (const float*)d_in[1];   // [4096,2,2048]
    const float* W1   = (const float*)d_in[2];   // [4096,2,4096]
    const float* W2   = (const float*)d_in[3];   // [4096,2,4096]
    const float* outW = (const float*)d_in[4];   // [3,4096,10]
    float* out = (float*)d_out;                  // [64,10]

    _Float16* bits = (_Float16*)d_ws;            // 64*2048 halves (256 KB)
    _Float16* acts = bits + (size_t)BATCH * FEAT * NUM_BITS; // 3*64*4096 halves

    const size_t layerElems = (size_t)BATCH * HID;

    eisani_encode<<<(BATCH * FEAT * NUM_BITS + 255) / 256, 256, 0, stream>>>(x, bits);

    // grid.x = (2H / 16 tiles) / 8 waves-per-block ; grid.y = 64/16 M-tiles
    dim3 lgrid(2 * HID / 16 / 8, BATCH / 16);
    eisani_layer<<<lgrid, 256, 0, stream>>>(bits, W0, acts,
                                            FEAT * NUM_BITS, HID);
    eisani_layer<<<lgrid, 256, 0, stream>>>(acts, W1, acts + layerElems,
                                            HID, HID);
    eisani_layer<<<lgrid, 256, 0, stream>>>(acts + layerElems, W2,
                                            acts + 2 * layerElems, HID, HID);

    eisani_out<<<dim3(NCLS, BATCH), 256, 0, stream>>>(acts, outW, out);
}
